// ParallelTransformerLayers_86895778332890
// MI455X (gfx1250) — compile-verified, matrix operand order checked by reading, not run on hardware
//
#include <hip/hip_runtime.h>

// ---------------------------------------------------------------------------
// Mamba block x2 (primary/secondary) for MI455X (gfx1250, wave32, WMMA)
// ---------------------------------------------------------------------------

typedef _Float16 half_t;
typedef __attribute__((ext_vector_type(16))) _Float16 v16h;
typedef __attribute__((ext_vector_type(8)))  float    v8f;

#define D_MODEL 1024
#define D_INNER 2048
#define D_STATE 16
#define D_CONV  4
#define DT_RANK 64
#define BB      2
#define LL      2048
#define ROWS    (BB * LL)                  // 4096
#define PROJ_W  (DT_RANK + 2 * D_STATE)    // 96

// ---------------------------------------------------------------------------
// f32 -> f16 elementwise convert (weights)
// ---------------------------------------------------------------------------
__global__ __launch_bounds__(256)
void convert_f16_kernel(const float* __restrict__ src, half_t* __restrict__ dst) {
    size_t i = (size_t)blockIdx.x * 256 + threadIdx.x;
    dst[i] = (half_t)src[i];
}

// extract first DT_RANK columns of proj (stride PROJ_W) as f16
__global__ __launch_bounds__(256)
void extract_dtr_kernel(const float* __restrict__ proj, half_t* __restrict__ dtr) {
    size_t i = (size_t)blockIdx.x * 256 + threadIdx.x;   // over ROWS*DT_RANK
    size_t r = i >> 6;           // /64
    size_t c = i & 63;
    dtr[i] = (half_t)proj[r * PROJ_W + c];
}

// ---------------------------------------------------------------------------
// LayerNorm over D_MODEL, output f16 (one block per row)
// ---------------------------------------------------------------------------
__global__ __launch_bounds__(256)
void layernorm_f16_kernel(const float* __restrict__ x,
                          const float* __restrict__ w,
                          const float* __restrict__ b,
                          half_t* __restrict__ out) {
    __shared__ float red[256];
    const int row = blockIdx.x;
    const float* xr = x + (size_t)row * D_MODEL;

    float s = 0.f;
    for (int c = threadIdx.x; c < D_MODEL; c += 256) s += xr[c];
    red[threadIdx.x] = s; __syncthreads();
    for (int off = 128; off; off >>= 1) {
        if (threadIdx.x < off) red[threadIdx.x] += red[threadIdx.x + off];
        __syncthreads();
    }
    const float mu = red[0] * (1.f / D_MODEL);
    __syncthreads();

    float v = 0.f;
    for (int c = threadIdx.x; c < D_MODEL; c += 256) { float t = xr[c] - mu; v += t * t; }
    red[threadIdx.x] = v; __syncthreads();
    for (int off = 128; off; off >>= 1) {
        if (threadIdx.x < off) red[threadIdx.x] += red[threadIdx.x + off];
        __syncthreads();
    }
    const float rstd = rsqrtf(red[0] * (1.f / D_MODEL) + 1e-5f);

    half_t* orow = out + (size_t)row * D_MODEL;
    for (int c = threadIdx.x; c < D_MODEL; c += 256)
        orow[c] = (half_t)((xr[c] - mu) * rstd * w[c] + b[c]);
}

// ---------------------------------------------------------------------------
// WMMA fragment helpers (layouts per CDNA5 ISA 7.12.2, wave32)
// ---------------------------------------------------------------------------
struct FragU { unsigned int u[8]; };

__device__ __forceinline__ FragU load_frag(const half_t* __restrict__ base,
                                           int k0, int hlf) {
    FragU f;
#pragma unroll
    for (int j = 0; j < 8; ++j) {
        const int kb = k0 + ((j >> 2) << 4) + (hlf << 3) + ((j & 3) << 1);
        f.u[j] = *(const unsigned int*)(base + kb);
    }
    return f;
}

__device__ __forceinline__ v16h frag_v(const FragU& f) {
    union { FragU f; v16h v; } c;
    c.f = f;
    return c.v;
}

// ---------------------------------------------------------------------------
// WMMA GEMM: C[M,N] (+residual) = A[M,K](f16,row-major) * W[N,K](f16,row-major)^T
// One wave computes a 16 x (16*NT) strip; A-fragment reused across NT wmmas.
// K-loop is software-pipelined: next iteration's fragments are issued before
// the current wmmas so global-load latency overlaps matrix math.
// ---------------------------------------------------------------------------
template<int NT>
__global__ __launch_bounds__(256)
void wmma_gemm_kernel(const half_t* __restrict__ A,
                      const half_t* __restrict__ W,
                      float* __restrict__ C,
                      const float* __restrict__ residual,
                      int M, int N, int K) {
    const int lane = threadIdx.x & 31;
    const int wib  = threadIdx.x >> 5;
    const int nStrips = N / (16 * NT);
    const long gwave = (long)blockIdx.x * 8 + wib;
    const int mt    = (int)(gwave / nStrips);
    const int strip = (int)(gwave % nStrips);
    const int m0 = mt * 16;
    const int n0 = strip * 16 * NT;
    if (m0 >= M) return;

    const int rc  = lane & 15;   // A row within tile == B column within tile
    const int hlf = lane >> 4;

    v8f acc[NT];
#pragma unroll
    for (int i = 0; i < NT; ++i) acc[i] = (v8f){0.f,0.f,0.f,0.f,0.f,0.f,0.f,0.f};

    const half_t* Arow = A + (size_t)(m0 + rc) * K;
    const half_t* Wrow[NT];
#pragma unroll
    for (int i = 0; i < NT; ++i) Wrow[i] = W + (size_t)(n0 + i * 16 + rc) * K;

    // prologue: fragments for k = 0
    FragU a_c = load_frag(Arow, 0, hlf);
    FragU b_c[NT];
#pragma unroll
    for (int i = 0; i < NT; ++i) b_c[i] = load_frag(Wrow[i], 0, hlf);

    for (int k0 = 0; k0 < K - 32; k0 += 32) {
        // issue next iteration's loads first (overlap with wmma below)
        FragU a_n = load_frag(Arow, k0 + 32, hlf);
        FragU b_n[NT];
#pragma unroll
        for (int i = 0; i < NT; ++i) b_n[i] = load_frag(Wrow[i], k0 + 32, hlf);

#pragma unroll
        for (int i = 0; i < NT; ++i)
            acc[i] = __builtin_amdgcn_wmma_f32_16x16x32_f16(
                false, frag_v(a_c), false, frag_v(b_c[i]), (short)0, acc[i],
                false, false);

        a_c = a_n;
#pragma unroll
        for (int i = 0; i < NT; ++i) b_c[i] = b_n[i];
    }

    // epilogue: last K block
#pragma unroll
    for (int i = 0; i < NT; ++i)
        acc[i] = __builtin_amdgcn_wmma_f32_16x16x32_f16(
            false, frag_v(a_c), false, frag_v(b_c[i]), (short)0, acc[i],
            false, false);

#pragma unroll
    for (int i = 0; i < NT; ++i) {
#pragma unroll
        for (int r = 0; r < 8; ++r) {
            const int m = m0 + r + hlf * 8;
            const int n = n0 + i * 16 + rc;
            const size_t idx = (size_t)m * N + n;
            float v = acc[i][r];
            if (residual) v += residual[idx];
            C[idx] = v;
        }
    }
}

// ---------------------------------------------------------------------------
// Causal depthwise conv1d (k=4) + SiLU; outputs f32 (for scan) and f16 (x_proj A)
// xz layout (B,L,2*D_INNER); xc = columns [0,D_INNER)
// ---------------------------------------------------------------------------
__global__ __launch_bounds__(256)
void conv_silu_kernel(const float* __restrict__ xz,
                      const float* __restrict__ cw,   // (D_INNER,4)
                      const float* __restrict__ cb,   // (D_INNER)
                      float* __restrict__ uf32,
                      half_t* __restrict__ uf16) {
    const size_t idx = (size_t)blockIdx.x * 256 + threadIdx.x;  // B*L*D_INNER
    const int d = (int)(idx & (D_INNER - 1));
    const int t = (int)((idx >> 11) & (LL - 1));
    const int b = (int)(idx >> 22);

    float acc = cb[d];
    const float* wv = cw + (size_t)d * D_CONV;
#pragma unroll
    for (int j = 0; j < D_CONV; ++j) {
        const int tt = t - (D_CONV - 1) + j;
        if (tt >= 0)
            acc += xz[((size_t)(b * LL + tt)) * (2 * D_INNER) + d] * wv[j];
    }
    const float u = acc * (1.f / (1.f + __expf(-acc)));   // SiLU
    uf32[idx] = u;
    uf16[idx] = (half_t)u;
}

// ---------------------------------------------------------------------------
// Selective scan, fused: softplus(dt) + recurrence + D-skip + SiLU gate -> f16
// 16 lanes per channel (one state each); y-reduction via shfl_xor butterfly.
// ---------------------------------------------------------------------------
__global__ __launch_bounds__(256)
void scan_kernel(const float* __restrict__ uf32,   // (B,L,D_INNER)
                 const float* __restrict__ dtraw,  // (B,L,D_INNER)
                 const float* __restrict__ proj,   // (B,L,96): [dtr|Bm|Cm]
                 const float* __restrict__ xz,     // (B,L,2*D_INNER): z at col D_INNER+d
                 const float* __restrict__ dt_b,   // (D_INNER)
                 const float* __restrict__ A_log,  // (D_INNER,D_STATE)
                 const float* __restrict__ Dp,     // (D_INNER)
                 half_t* __restrict__ yg) {        // (B,L,D_INNER) f16, gated
    const int s    = threadIdx.x & 15;
    const int grp  = threadIdx.x >> 4;                     // 16 channels / block
    const int blkC = blockIdx.x & (D_INNER / 16 - 1);      // 128 blocks per batch
    const int b    = blockIdx.x >> 7;
    const int d    = blkC * 16 + grp;

    const float A_s = -__expf(A_log[(size_t)d * D_STATE + s]);
    const float dtb = dt_b[d];
    const float Dd  = Dp[d];

    float h = 0.f;
    const size_t baseBL = (size_t)b * LL;
    for (int t = 0; t < LL; ++t) {
        const size_t rowu = (baseBL + t) * D_INNER + d;
        const float u = uf32[rowu];
        const float x = dtraw[rowu] + dtb;
        const float dt = (x > 20.f) ? x : log1pf(__expf(x));  // softplus

        const size_t prow = (baseBL + t) * PROJ_W;
        const float Bv = proj[prow + DT_RANK + s];
        const float Cv = proj[prow + DT_RANK + D_STATE + s];

        const float dA = __expf(dt * A_s);
        h = dA * h + (dt * u) * Bv;
        float y = h * Cv;
        y += __shfl_xor(y, 1, 16);
        y += __shfl_xor(y, 2, 16);
        y += __shfl_xor(y, 4, 16);
        y += __shfl_xor(y, 8, 16);

        if (s == 0) {
            const float z = xz[(baseBL + t) * (2 * D_INNER) + D_INNER + d];
            const float g = z * (1.f / (1.f + __expf(-z)));
            yg[rowu] = (half_t)((y + u * Dd) * g);
        }
    }
}

// ---------------------------------------------------------------------------
// Host orchestration
// ---------------------------------------------------------------------------
extern "C" void kernel_launch(void* const* d_in, const int* in_sizes, int n_in,
                              void* d_out, int out_size, void* d_ws, size_t ws_size,
                              hipStream_t stream) {
    (void)in_sizes; (void)n_in; (void)out_size; (void)ws_size;

    const float* x_primary   = (const float*)d_in[0];
    const float* x_secondary = (const float*)d_in[1];

    char* ws = (char*)d_ws;
    size_t off = 0;
    auto alloc = [&](size_t bytes) -> void* {
        void* p = ws + off;
        off = (off + bytes + 255) & ~(size_t)255;
        return p;
    };

    half_t* HF16   = (half_t*)alloc((size_t)ROWS * D_MODEL * 2);
    half_t* WIN16  = (half_t*)alloc((size_t)(2 * D_INNER) * D_MODEL * 2);
    float*  XZ     = (float*) alloc((size_t)ROWS * (2 * D_INNER) * 4);
    float*  UF32   = (float*) alloc((size_t)ROWS * D_INNER * 4);
    half_t* UF16   = (half_t*)alloc((size_t)ROWS * D_INNER * 2);
    half_t* WXP16  = (half_t*)alloc((size_t)PROJ_W * D_INNER * 2);
    float*  PROJ   = (float*) alloc((size_t)ROWS * PROJ_W * 4);
    half_t* DTR16  = (half_t*)alloc((size_t)ROWS * DT_RANK * 2);
    half_t* WDT16  = (half_t*)alloc((size_t)D_INNER * DT_RANK * 2);
    float*  DTRAW  = (float*) alloc((size_t)ROWS * D_INNER * 4);
    half_t* YG16   = (half_t*)alloc((size_t)ROWS * D_INNER * 2);
    half_t* WOUT16 = (half_t*)alloc((size_t)D_MODEL * D_INNER * 2);

    for (int pass = 0; pass < 2; ++pass) {
        const int base = 2 + pass * 11;
        const float* ln_w   = (const float*)d_in[base + 0];
        const float* ln_b   = (const float*)d_in[base + 1];
        const float* inprj  = (const float*)d_in[base + 2];
        const float* conv_w = (const float*)d_in[base + 3];
        const float* conv_b = (const float*)d_in[base + 4];
        const float* x_proj = (const float*)d_in[base + 5];
        const float* dt_w   = (const float*)d_in[base + 6];
        const float* dt_b   = (const float*)d_in[base + 7];
        const float* A_log  = (const float*)d_in[base + 8];
        const float* Dpar   = (const float*)d_in[base + 9];
        const float* outprj = (const float*)d_in[base + 10];

        const float* xin = pass ? x_secondary : x_primary;
        float* outp = (float*)d_out + (size_t)pass * ROWS * D_MODEL;
        const float* residual = pass ? x_primary : nullptr;

        // 1) weights -> f16
        convert_f16_kernel<<<(2 * D_INNER) * D_MODEL / 256, 256, 0, stream>>>(inprj, WIN16);
        convert_f16_kernel<<<PROJ_W * D_INNER / 256, 256, 0, stream>>>(x_proj, WXP16);
        convert_f16_kernel<<<D_INNER * DT_RANK / 256, 256, 0, stream>>>(dt_w, WDT16);
        convert_f16_kernel<<<D_MODEL * D_INNER / 256, 256, 0, stream>>>(outprj, WOUT16);

        // 2) layernorm -> f16
        layernorm_f16_kernel<<<ROWS, 256, 0, stream>>>(xin, ln_w, ln_b, HF16);

        // 3) in_proj: (ROWS x 2*D_INNER) = HF16 * WIN16^T
        wmma_gemm_kernel<4><<<(ROWS / 16) * ((2 * D_INNER) / 64) / 8, 256, 0, stream>>>(
            HF16, WIN16, XZ, nullptr, ROWS, 2 * D_INNER, D_MODEL);

        // 4) causal conv + SiLU
        conv_silu_kernel<<<(size_t)ROWS * D_INNER / 256, 256, 0, stream>>>(
            XZ, conv_w, conv_b, UF32, UF16);

        // 5) x_proj: (ROWS x 96) = UF16 * WXP16^T   (N=96 -> 32-wide strips)
        wmma_gemm_kernel<2><<<(ROWS / 16) * (PROJ_W / 32) / 8, 256, 0, stream>>>(
            UF16, WXP16, PROJ, nullptr, ROWS, PROJ_W, D_INNER);

        // 6) dtr f32 (strided in PROJ) -> f16
        extract_dtr_kernel<<<ROWS * DT_RANK / 256, 256, 0, stream>>>(PROJ, DTR16);

        // 7) dt_proj: (ROWS x D_INNER) = DTR16 * WDT16^T
        wmma_gemm_kernel<4><<<(ROWS / 16) * (D_INNER / 64) / 8, 256, 0, stream>>>(
            DTR16, WDT16, DTRAW, nullptr, ROWS, D_INNER, DT_RANK);

        // 8) fused softplus + selective scan + skip + gate -> f16
        scan_kernel<<<BB * (D_INNER / 16), 256, 0, stream>>>(
            UF32, DTRAW, PROJ, XZ, dt_b, A_log, Dpar, YG16);

        // 9) out_proj (+ residual for secondary), straight into d_out
        wmma_gemm_kernel<4><<<(ROWS / 16) * (D_MODEL / 64) / 8, 256, 0, stream>>>(
            YG16, WOUT16, outp, residual, ROWS, D_MODEL, D_INNER);
    }
}
